// MinifloatLinearPTQ_90718299226752
// MI455X (gfx1250) — 8-bit matrix operand order repaired and checked by reading, compiled, not run on hardware
//
#include <hip/hip_runtime.h>
#include <hip/hip_bf16.h>


__device__ __forceinline__ unsigned int fp8_e4m3_byte(float f) {
    unsigned int u   = __float_as_uint(f);
    unsigned int sgn = (u >> 24) & 0x80u;
    float af = fminf(fabsf(f), 448.0f);
    unsigned int a = __float_as_uint(af);
    int e32 = (int)(a >> 23) - 127;
    unsigned int mant = a & 0x7FFFFFu;
    unsigned int keep = mant >> 20;
    unsigned int rest = mant & 0xFFFFFu;
    unsigned int rnd  = (unsigned int)((rest > 0x80000u) ||
                                       ((rest == 0x80000u) && (keep & 1u)));
    unsigned int codeN = ((((unsigned int)(e32 + 7)) << 3) | keep) + rnd;
    codeN = codeN > 0x7Eu ? 0x7Eu : codeN;
    unsigned int codeS = (unsigned int)rintf(af * 512.0f);
    unsigned int code  = (af < 0.015625f) ? codeS : codeN;
    return sgn | code;
}

__device__ __forceinline__ float quant_e4m3_val(float f) {
    unsigned int c = fp8_e4m3_byte(f);
    unsigned int e = (c >> 3) & 0xFu;
    unsigned int m = c & 7u;
    float mag = (e == 0u)
        ? ((float)m * 0.001953125f)
        : __uint_as_float((e + 120u) << 23) * (1.0f + (float)m * 0.125f);
    return (c & 0x80u) ? -mag : mag;
}

__global__ __launch_bounds__(256) void quant_e4m3_kernel(
    const float* __restrict__ src, unsigned char* __restrict__ dst, int n8) {
    int i = blockIdx.x * blockDim.x + threadIdx.x;
    if (i >= n8) return;
    const float4* s = (const float4*)src;
    float4 f0 = s[2 * i];
    float4 f1 = s[2 * i + 1];
    unsigned int lo = fp8_e4m3_byte(f0.x)        | (fp8_e4m3_byte(f0.y) << 8) |
                      (fp8_e4m3_byte(f0.z) << 16) | (fp8_e4m3_byte(f0.w) << 24);
    unsigned int hi = fp8_e4m3_byte(f1.x)        | (fp8_e4m3_byte(f1.y) << 8) |
                      (fp8_e4m3_byte(f1.z) << 16) | (fp8_e4m3_byte(f1.w) << 24);
    ((uint2*)dst)[i] = make_uint2(lo, hi);
}

typedef __attribute__((ext_vector_type(16))) int   v16i;
typedef __attribute__((ext_vector_type(8)))  float v8f;

__global__ __launch_bounds__(256) void gemm_fp8_wmma_kernel(
    const unsigned char* __restrict__ xq,
    const unsigned char* __restrict__ wq,
    const float* __restrict__ bias,
    float* __restrict__ out)
{
    constexpr int K = 1024, N = 1024;
    const int lane = threadIdx.x & 31;
    const int wave = threadIdx.x >> 5;
    const int wm   = wave & 3;
    const int wn   = wave >> 2;
    const int l15  = lane & 15;
    const int hi   = lane >> 4;

    const int m0 = blockIdx.y * 128 + wm * 32;
    const int n0 = blockIdx.x * 64  + wn * 32;

    const unsigned char* pa0 = xq + (size_t)(m0 + l15) * K;
    const unsigned char* pa1 = pa0 + (size_t)16 * K;
    const unsigned char* pb0 = wq + (size_t)(n0 + l15) * K;
    const unsigned char* pb1 = pb0 + (size_t)16 * K;

    float b0 = quant_e4m3_val(bias[n0 + l15]);
    float b1 = quant_e4m3_val(bias[n0 + 16 + l15]);
    v8f acc00, acc01, acc10, acc11;
#pragma unroll
    for (int j = 0; j < 8; ++j) {
        acc00[j] = b0; acc10[j] = b0;
        acc01[j] = b1; acc11[j] = b1;
    }

    const int aoff = hi * 8;
    const int boff = hi * 16;

    for (int k0 = 0; k0 < K; k0 += 128) {
        v16i A0, A1, B0, B1;
#pragma unroll
        for (int p = 0; p < 8; ++p) {
            const int ko = k0 + (p >> 1) * 32 + (p & 1) * 8 + boff;
            unsigned long long d0 = *(const unsigned long long*)(pa0 + ko);
            unsigned long long d1 = *(const unsigned long long*)(pa1 + ko);
            A0[2 * p]     = (int)(unsigned int)d0;
            A0[2 * p + 1] = (int)(unsigned int)(d0 >> 32);
            A1[2 * p]     = (int)(unsigned int)d1;
            A1[2 * p + 1] = (int)(unsigned int)(d1 >> 32);
        }
#pragma unroll
        for (int q = 0; q < 4; ++q) {
            const int ko = k0 + q * 32 + boff;
            uint4 d0 = *(const uint4*)(pb0 + ko);
            uint4 d1 = *(const uint4*)(pb1 + ko);
            B0[4 * q + 0] = (int)d0.x; B0[4 * q + 1] = (int)d0.y;
            B0[4 * q + 2] = (int)d0.z; B0[4 * q + 3] = (int)d0.w;
            B1[4 * q + 0] = (int)d1.x; B1[4 * q + 1] = (int)d1.y;
            B1[4 * q + 2] = (int)d1.z; B1[4 * q + 3] = (int)d1.w;
        }
        acc00 = __builtin_amdgcn_wmma_f32_16x16x128_fp8_fp8(A0, B0, (short)0, acc00, false, false);
        acc01 = __builtin_amdgcn_wmma_f32_16x16x128_fp8_fp8(A0, B1, (short)0, acc01, false, false);
        acc10 = __builtin_amdgcn_wmma_f32_16x16x128_fp8_fp8(A1, B0, (short)0, acc10, false, false);
        acc11 = __builtin_amdgcn_wmma_f32_16x16x128_fp8_fp8(A1, B1, (short)0, acc11, false, false);
    }

    float* o = out + (size_t)(m0 + hi * 8) * N + n0 + l15;
#pragma unroll
    for (int j = 0; j < 8; ++j) {
        o[(size_t)j * N]             = acc00[j];
        o[(size_t)j * N + 16]        = acc01[j];
        o[(size_t)(j + 16) * N]      = acc10[j];
        o[(size_t)(j + 16) * N + 16] = acc11[j];
    }
}

extern "C" void kernel_launch(void* const* d_in, const int* in_sizes, int n_in,
                              void* d_out, int out_size, void* d_ws, size_t ws_size,
                              hipStream_t stream) {
    (void)in_sizes; (void)n_in; (void)out_size; (void)ws_size;
    const float* x = (const float*)d_in[0];
    const float* w = (const float*)d_in[1];
    const float* b = (const float*)d_in[2];
    float* out = (float*)d_out;

    constexpr int TOK = 32768, DIN = 1024, DOUT = 1024;
    unsigned char* xq = (unsigned char*)d_ws;
    unsigned char* wq = xq + (size_t)TOK * DIN;

    const int nx8 = TOK * DIN / 8;
    const int nw8 = DOUT * DIN / 8;
    quant_e4m3_kernel<<<nx8 / 256, 256, 0, stream>>>(x, xq, nx8);
    quant_e4m3_kernel<<<nw8 / 256, 256, 0, stream>>>(w, wq, nw8);

    dim3 grid(DOUT / 64, TOK / 128);
    gemm_fp8_wmma_kernel<<<grid, dim3(256), 0, stream>>>(xq, wq, b, out);
}
